// MoeModule_26611617366087
// MI455X (gfx1250) — matrix operand order repaired and checked by reading, not compiled, not run on hardware
//
#include <hip/hip_runtime.h>

#define S_TOK   8192
#define DMODEL  1024
#define DFF     4096
#define NEXP    16
#define CAPC    640   // floor(1.25*8192/16)=640, even, >=4

// Plain POD vector types (avoid HIP class-type vectors inside bit-casts/unions)
struct __align__(16) U128 { unsigned int x, y, z, w; };
struct __align__(32) U256 { U128 lo, hi; };
struct __align__(8)  US4  { unsigned short x, y, z, w; };
struct __align__(16) F128 { float x, y, z, w; };

typedef __attribute__((ext_vector_type(16))) __bf16 v16bf;
typedef __attribute__((ext_vector_type(8)))  float  v8f;

__device__ __forceinline__ unsigned short f2bf(float f) {
  unsigned int u = __float_as_uint(f);
  unsigned int r = u + 0x7FFFu + ((u >> 16) & 1u);   // round-to-nearest-even
  return (unsigned short)(r >> 16);
}

__device__ __forceinline__ float gelu_tanh(float x) {
  float x3 = x * x * x;
  return 0.5f * x * (1.0f + tanhf(0.7978845608028654f * (x + 0.044715f * x3)));
}

// CDNA5 async copy: global -> LDS, 16B per lane, tracked by ASYNCcnt.
// lds_byte = LDS byte address (low 32 bits of flat shared pointer == LDS offset).
__device__ __forceinline__ void async_cp_b128(unsigned lds_byte, const void* gaddr) {
  asm volatile("global_load_async_to_lds_b128 %0, %1, off"
               :: "v"(lds_byte), "v"(gaddr) : "memory");
}
__device__ __forceinline__ void wait_async0() {
  asm volatile("s_wait_asynccnt 0x0" ::: "memory");
}

// ---------------- weight convert + transpose: W[K][N] f32 -> WT[N][K] bf16 ----
__global__ void convt_kernel(const float* __restrict__ W, unsigned short* __restrict__ WT,
                             int K, int N) {
  __shared__ float tile[32][33];
  const int e  = blockIdx.z;
  const int k0 = blockIdx.y * 32;
  const int n0 = blockIdx.x * 32;
  const float* src = W + (size_t)e * K * N;
  unsigned short* dst = WT + (size_t)e * N * K;
  const int t  = threadIdx.x;           // 256 threads
  const int r  = t >> 3;                // 0..31
  const int c4 = (t & 7) * 4;           // 0..28
  F128 v = *(const F128*)(src + (size_t)(k0 + r) * N + n0 + c4);
  tile[r][c4 + 0] = v.x; tile[r][c4 + 1] = v.y;
  tile[r][c4 + 2] = v.z; tile[r][c4 + 3] = v.w;
  __syncthreads();
  US4 o;
  o.x = f2bf(tile[c4 + 0][r]);
  o.y = f2bf(tile[c4 + 1][r]);
  o.z = f2bf(tile[c4 + 2][r]);
  o.w = f2bf(tile[c4 + 3][r]);
  *(US4*)(dst + (size_t)(n0 + r) * K + k0 + c4) = o;
}

// ---------------- gate: one wave per token, 16 logits + softmax top1 ----------
__global__ void gate_kernel(const float* __restrict__ x, const float* __restrict__ gw,
                            int* __restrict__ top1, float* __restrict__ prob) {
  const int wave = threadIdx.x >> 5;
  const int lane = threadIdx.x & 31;
  const int t = blockIdx.x * 4 + wave;
  const float* tok = x + (size_t)t * DMODEL;
  float xr[32];
#pragma unroll
  for (int i = 0; i < 32; ++i) xr[i] = tok[i * 32 + lane];

  float mylogit = 0.0f;
  for (int e = 0; e < NEXP; ++e) {
    const float* g = gw + (size_t)e * DMODEL;
    float acc = 0.0f;
#pragma unroll
    for (int i = 0; i < 32; ++i) acc += xr[i] * g[i * 32 + lane];
#pragma unroll
    for (int m = 16; m >= 1; m >>= 1) acc += __shfl_xor(acc, m, 32);
    if (lane == e) mylogit = acc;
  }
  // argmax + max over lanes 0..15 (group closed under xor masks 1,2,4,8)
  float v = (lane < 16) ? mylogit : -3.4e38f;
  int idx  = (lane < 16) ? lane : 999;
#pragma unroll
  for (int m = 8; m >= 1; m >>= 1) {
    float ov = __shfl_xor(v, m, 32);
    int   oi = __shfl_xor(idx, m, 32);
    if (ov > v || (ov == v && oi < idx)) { v = ov; idx = oi; }
  }
  float ex = (lane < 16) ? __expf(mylogit - v) : 0.0f;
#pragma unroll
  for (int m = 8; m >= 1; m >>= 1) ex += __shfl_xor(ex, m, 32);
  if (lane == 0) {
    top1[t] = idx;
    prob[t] = 1.0f / ex;   // exp(max - max)/sum == prob of the argmax expert
  }
}

// ---------------- router: 16 lanes, sequential capacity scan ------------------
__global__ void route_kernel(const int* __restrict__ top1, int* __restrict__ slot_token,
                             int* __restrict__ token_slot) {
  const int e = threadIdx.x;
  if (e >= NEXP) return;
  int cnt = 0;
  for (int t = 0; t < S_TOK; ++t) {
    if (top1[t] == e) {
      if (cnt < CAPC) {
        slot_token[e * CAPC + cnt] = t;
        token_slot[t] = e * CAPC + cnt;
      } else {
        token_slot[t] = -1;           // overflow token dropped
      }
      ++cnt;
    }
  }
  for (int c = cnt; c < CAPC; ++c) slot_token[e * CAPC + c] = -1;
}

// ---------------- dispatch: gather tokens -> bf16 [E][CAP][D] -----------------
__global__ void dispatch_kernel(const float* __restrict__ x, const int* __restrict__ slot_token,
                                unsigned short* __restrict__ disp) {
  const int slot = blockIdx.x;
  const int t = slot_token[slot];
  const int c = threadIdx.x * 4;
  US4 o;
  if (t >= 0) {
    const float* src = x + (size_t)t * DMODEL + c;
    o.x = f2bf(src[0]); o.y = f2bf(src[1]); o.z = f2bf(src[2]); o.w = f2bf(src[3]);
  } else {
    o.x = o.y = o.z = o.w = 0;
  }
  *(US4*)(disp + (size_t)slot * DMODEL + c) = o;
}

// ---------------- tiled bf16 WMMA GEMM: C[640xN] = A[640xK] * BT[NxK]^T -------
// Block tile 128(M) x 128(N); 8 waves, each 32x64 (2 A-frags x 4 B-frags = 8 WMMA).
// K staged 32-wide via double-buffered LDS filled with async global->LDS copies.
template <int KTOT, bool GELU, bool OUT_BF16>
__global__ void gemm_wmma(const unsigned short* __restrict__ A,
                          const unsigned short* __restrict__ BT,
                          void* __restrict__ Cout, int N) {
  // row stride 40 shorts (80B, 16B-aligned) to stagger LDS banks
  __shared__ __align__(16) unsigned short As[2][128 * 40];
  __shared__ __align__(16) unsigned short Bs[2][128 * 40];

  const int e  = blockIdx.z;
  const int m0 = blockIdx.y * 128;
  const int n0 = blockIdx.x * 128;
  const unsigned short* Ae = A  + (size_t)e * CAPC * KTOT;
  const unsigned short* Be = BT + (size_t)e * (size_t)N * KTOT;

  const int tid  = threadIdx.x;
  const int lane = tid & 31;
  const int wave = tid >> 5;
  const int wm = wave & 3;        // 0..3 -> 32-row M sub-tile
  const int wn = wave >> 2;       // 0..1 -> 64-col N half
  const int lm = lane & 15;
  const int kb = lane >> 4;

  v8f acc[2][4];
#pragma unroll
  for (int a = 0; a < 2; ++a)
#pragma unroll
    for (int i = 0; i < 4; ++i)
#pragma unroll
      for (int j = 0; j < 8; ++j) acc[a][i][j] = 0.0f;

  // LDS byte bases of the two buffers (low 32 bits of flat addr == LDS offset)
  const unsigned aB[2] = { (unsigned)(size_t)&As[0][0], (unsigned)(size_t)&As[1][0] };
  const unsigned bB[2] = { (unsigned)(size_t)&Bs[0][0], (unsigned)(size_t)&Bs[1][0] };

  // per-thread staging coords: 512 16B-chunks per tile, 2 per thread
  const int r0 = tid >> 2,          c80 = (tid & 3) * 8;
  const int r1 = (tid + 256) >> 2,  c81 = ((tid + 256) & 3) * 8;

  auto stage = [&](int k0, int buf) {
    async_cp_b128(aB[buf] + (unsigned)(r0 * 40 + c80) * 2,
                  Ae + (size_t)(m0 + r0) * KTOT + k0 + c80);
    async_cp_b128(aB[buf] + (unsigned)(r1 * 40 + c81) * 2,
                  Ae + (size_t)(m0 + r1) * KTOT + k0 + c81);
    async_cp_b128(bB[buf] + (unsigned)(r0 * 40 + c80) * 2,
                  Be + (size_t)(n0 + r0) * KTOT + k0 + c80);
    async_cp_b128(bB[buf] + (unsigned)(r1 * 40 + c81) * 2,
                  Be + (size_t)(n0 + r1) * KTOT + k0 + c81);
  };

  stage(0, 0);
  int buf = 0;
  for (int k0 = 0; k0 < KTOT; k0 += 32) {
    wait_async0();        // this K-slab is resident in LDS
    __syncthreads();      // all waves arrived => everyone's frag reads of the
                          // other buffer (prev iter) are complete
    if (k0 + 32 < KTOT) stage(k0 + 32, buf ^ 1);   // overlap with WMMAs below

    const unsigned short* Asp = As[buf];
    const unsigned short* Bsp = Bs[buf];

    // A fragments: lane half kb holds K in [8kb,8kb+8) and [16+8kb,16+8kb+8)
    v16bf af[2];
#pragma unroll
    for (int a = 0; a < 2; ++a) {
      const int a_lo = (wm * 32 + a * 16 + lm) * 40 + kb * 8;
      U256 au;
      au.lo = *(const U128*)(Asp + a_lo);
      au.hi = *(const U128*)(Asp + a_lo + 16);
      af[a] = __builtin_bit_cast(v16bf, au);
    }
#pragma unroll
    for (int nt = 0; nt < 4; ++nt) {
      const int b_lo = (wn * 64 + nt * 16 + lm) * 40 + kb * 8;
      U256 bu;
      bu.lo = *(const U128*)(Bsp + b_lo);
      bu.hi = *(const U128*)(Bsp + b_lo + 16);
      v16bf bfr = __builtin_bit_cast(v16bf, bu);
      acc[0][nt] = __builtin_amdgcn_wmma_f32_16x16x32_bf16(
          false, af[0], false, bfr, (short)0, acc[0][nt], false, false);
      acc[1][nt] = __builtin_amdgcn_wmma_f32_16x16x32_bf16(
          false, af[1], false, bfr, (short)0, acc[1][nt], false, false);
    }
    buf ^= 1;
  }

  // epilogue: C VGPR r -> row 8*kb + r, col = lane%16
#pragma unroll
  for (int a = 0; a < 2; ++a)
#pragma unroll
    for (int nt = 0; nt < 4; ++nt) {
      const int col = n0 + wn * 64 + nt * 16 + lm;
#pragma unroll
      for (int r = 0; r < 8; ++r) {
        const int row = m0 + wm * 32 + a * 16 + kb * 8 + r;
        float v = acc[a][nt][r];
        if (GELU) v = gelu_tanh(v);
        const size_t off = (size_t)e * CAPC * N + (size_t)row * N + col;
        if (OUT_BF16) ((unsigned short*)Cout)[off] = f2bf(v);
        else          ((float*)Cout)[off] = v;
      }
    }
}

// ---------------- combine: out[t] = prob * expert_out[slot], 0 if dropped -----
__global__ void combine_kernel(const int* __restrict__ token_slot, const float* __restrict__ prob,
                               const float* __restrict__ eout, float* __restrict__ out) {
  const int t = blockIdx.x;
  const int c = threadIdx.x * 4;
  const int slot = token_slot[t];
  const float w = prob[t];
  F128 o;
  if (slot >= 0) {
    const float* src = eout + (size_t)slot * DMODEL + c;
    o.x = w * src[0]; o.y = w * src[1]; o.z = w * src[2]; o.w = w * src[3];
  } else {
    o.x = o.y = o.z = o.w = 0.0f;
  }
  *(F128*)(out + (size_t)t * DMODEL + c) = o;
}

extern "C" void kernel_launch(void* const* d_in, const int* in_sizes, int n_in,
                              void* d_out, int out_size, void* d_ws, size_t ws_size,
                              hipStream_t stream) {
  (void)in_sizes; (void)n_in; (void)out_size; (void)ws_size;
  const float* x  = (const float*)d_in[0];   // [4,2048,1024] -> [8192,1024]
  const float* gw = (const float*)d_in[1];   // [16,1024]
  const float* w1 = (const float*)d_in[2];   // [16,1024,4096]
  const float* w2 = (const float*)d_in[3];   // [16,4096,1024]
  float* out = (float*)d_out;

  // workspace carve-up (~416 MB total)
  char* p = (char*)d_ws;
  auto take = [&](size_t bytes) { char* q = p; p += (bytes + 255) & ~(size_t)255; return q; };
  unsigned short* w1T  = (unsigned short*)take((size_t)NEXP * DFF * DMODEL * 2);  // [E][N=FF][K=D]
  unsigned short* w2T  = (unsigned short*)take((size_t)NEXP * DMODEL * DFF * 2);  // [E][N=D][K=FF]
  unsigned short* disp = (unsigned short*)take((size_t)NEXP * CAPC * DMODEL * 2);
  unsigned short* hbuf = (unsigned short*)take((size_t)NEXP * CAPC * DFF * 2);
  float* eout          = (float*)take((size_t)NEXP * CAPC * DMODEL * 4);
  int*   top1          = (int*)take((size_t)S_TOK * 4);
  float* prob          = (float*)take((size_t)S_TOK * 4);
  int*   slot_token    = (int*)take((size_t)NEXP * CAPC * 4);
  int*   token_slot    = (int*)take((size_t)S_TOK * 4);

  // 1) weights: fp32 -> bf16, transposed to [N][K]
  convt_kernel<<<dim3(DFF / 32, DMODEL / 32, NEXP), 256, 0, stream>>>(w1, w1T, DMODEL, DFF);
  convt_kernel<<<dim3(DMODEL / 32, DFF / 32, NEXP), 256, 0, stream>>>(w2, w2T, DFF, DMODEL);
  // 2) gating (one wave per token)
  gate_kernel<<<S_TOK / 4, 128, 0, stream>>>(x, gw, top1, prob);
  // 3) capacity routing
  route_kernel<<<1, 32, 0, stream>>>(top1, slot_token, token_slot);
  // 4) dispatch gather -> bf16
  dispatch_kernel<<<NEXP * CAPC, 256, 0, stream>>>(x, slot_token, disp);
  // 5) expert FFN: GEMM1 + GELU (bf16 out), then GEMM2 (fp32 out)
  gemm_wmma<DMODEL, true, true><<<dim3(DFF / 128, CAPC / 128, NEXP), 256, 0, stream>>>(
      disp, w1T, (void*)hbuf, DFF);
  gemm_wmma<DFF, false, false><<<dim3(DMODEL / 128, CAPC / 128, NEXP), 256, 0, stream>>>(
      hbuf, w2T, (void*)eout, DMODEL);
  // 6) weighted combine back to token order
  combine_kernel<<<S_TOK, 256, 0, stream>>>(token_slot, prob, eout, out);
}
